// BIAN_56650618634654
// MI455X (gfx1250) — matrix operand order repaired and checked by reading, not compiled, not run on hardware
//
#include <hip/hip_runtime.h>
#include <math.h>

// ---------------------------------------------------------------------------
// GNN (SAGE + GAT + MLP) for MI455X / gfx1250, wave32.
//  - GEMMs: v_wmma_f32_16x16x32_bf16, weights async-staged to LDS
//    (global_load_async_to_lds_b128 / s_wait_asynccnt) and pre-swizzled into
//    per-lane B-fragment order so each fragment is 2x ds_load_b128.
//  - Graph aggregation / attention softmax: f32 atomics resolving in the
//    192MB L2 (all segment buffers fit), float4-granular gathers.
// ---------------------------------------------------------------------------

typedef __attribute__((ext_vector_type(16))) __bf16 v16bf;
typedef __attribute__((ext_vector_type(8)))  float  v8f;

constexpr int NN   = 50000;          // nodes
constexpr int NE   = 800000;         // edges
constexpr int NTOT = NN + NE;        // edges incl. self-loops (GAT)

static __device__ __forceinline__ float lrelu(float v) { return v > 0.f ? v : 0.2f * v; }

// float atomic max via sign-split integer atomics (init with -3e38).
static __device__ __forceinline__ void atomicMaxF(float* a, float v) {
  if (v >= 0.f) atomicMax((int*)a, __float_as_int(v));
  else          atomicMin((unsigned int*)a, (unsigned int)__float_as_int(v));
}

// ---------------------------------------------------------------------------
// elementwise / graph kernels
// ---------------------------------------------------------------------------

__global__ void gnn_fill(float* __restrict__ p, float v, int n) {
  int i = blockIdx.x * blockDim.x + threadIdx.x;
  if (i < n) p[i] = v;
}

__global__ void gnn_deg(const int* __restrict__ dst, float* __restrict__ cnt, int e) {
  int i = blockIdx.x * blockDim.x + threadIdx.x;
  if (i < e) atomicAdd(cnt + dst[i], 1.f);
}

// scatter-add of 64-wide f32 rows: one thread per (edge, float4-chunk)
__global__ void gnn_scatter64(const float* __restrict__ x, const int* __restrict__ src,
                              const int* __restrict__ dst, float* __restrict__ out, int e) {
  int i = blockIdx.x * blockDim.x + threadIdx.x;
  if (i >= e * 16) return;
  int eid = i >> 4, c4 = i & 15;
  int s = src[eid], d = dst[eid];
  const float4 v = ((const float4*)x)[(size_t)s * 16 + c4];
  float* o = out + (size_t)d * 64 + c4 * 4;
  atomicAdd(o + 0, v.x); atomicAdd(o + 1, v.y);
  atomicAdd(o + 2, v.z); atomicAdd(o + 3, v.w);
}

__global__ void gnn_meandiv(float* __restrict__ s, const float* __restrict__ cnt, int n) {
  int i = blockIdx.x * blockDim.x + threadIdx.x;
  if (i < n) s[i] /= fmaxf(cnt[i >> 6], 1.f);
}

// per-channel batch mean/var (biased, like jnp.var), one block per channel
__global__ void gnn_colstats(const float* __restrict__ x, int C,
                             float* __restrict__ mean, float* __restrict__ var) {
  int c = blockIdx.x;
  __shared__ float ss[256], sq[256];
  float s = 0.f, q = 0.f;
  for (int r = threadIdx.x; r < NN; r += 256) {
    float v = x[(size_t)r * C + c];
    s += v; q += v * v;
  }
  ss[threadIdx.x] = s; sq[threadIdx.x] = q;
  __syncthreads();
  for (int st = 128; st > 0; st >>= 1) {
    if (threadIdx.x < st) { ss[threadIdx.x] += ss[threadIdx.x + st]; sq[threadIdx.x] += sq[threadIdx.x + st]; }
    __syncthreads();
  }
  if (threadIdx.x == 0) {
    float m = ss[0] / NN;
    mean[c] = m;
    var[c]  = sq[0] / NN - m * m;
  }
}

__global__ void gnn_bnrelu(float* __restrict__ x, const float* __restrict__ mean,
                           const float* __restrict__ var, const float* __restrict__ g,
                           const float* __restrict__ be, int C, int total) {
  int i = blockIdx.x * blockDim.x + threadIdx.x;
  if (i >= total) return;
  int c = i % C;
  float v = (x[i] - mean[c]) * rsqrtf(var[c] + 1e-5f) * g[c] + be[c];
  x[i] = fmaxf(v, 0.f);
}

// ---------------------------------------------------------------------------
// GAT attention kernels (H heads, Dh = 64)
// ---------------------------------------------------------------------------

__global__ void gnn_logits(const float* __restrict__ h, const float* __restrict__ as,
                           const float* __restrict__ ad, float* __restrict__ es,
                           float* __restrict__ ed, int H) {
  int i = blockIdx.x * blockDim.x + threadIdx.x;
  if (i >= NN * H) return;
  int node = i / H, hh = i % H;
  const float* r  = h  + (size_t)node * H * 64 + hh * 64;
  const float* a1 = as + hh * 64;
  const float* a2 = ad + hh * 64;
  float e1 = 0.f, e2 = 0.f;
  for (int d = 0; d < 64; ++d) { e1 += r[d] * a1[d]; e2 += r[d] * a2[d]; }
  es[i] = e1; ed[i] = e2;
}

__global__ void gnn_attmax(const int* __restrict__ src, const int* __restrict__ dst,
                           const float* __restrict__ es, const float* __restrict__ ed,
                           float* __restrict__ mx, int H) {
  int i = blockIdx.x * blockDim.x + threadIdx.x;
  if (i >= NTOT * H) return;
  int e = i / H, hh = i % H;
  int s, d;
  if (e < NE) { s = src[e]; d = dst[e]; } else { s = d = e - NE; }
  atomicMaxF(mx + (size_t)d * H + hh, lrelu(es[(size_t)s * H + hh] + ed[(size_t)d * H + hh]));
}

__global__ void gnn_attsum(const int* __restrict__ src, const int* __restrict__ dst,
                           const float* __restrict__ es, const float* __restrict__ ed,
                           const float* __restrict__ mx, float* __restrict__ sm, int H) {
  int i = blockIdx.x * blockDim.x + threadIdx.x;
  if (i >= NTOT * H) return;
  int e = i / H, hh = i % H;
  int s, d;
  if (e < NE) { s = src[e]; d = dst[e]; } else { s = d = e - NE; }
  float v = lrelu(es[(size_t)s * H + hh] + ed[(size_t)d * H + hh]);
  atomicAdd(sm + (size_t)d * H + hh, expf(v - mx[(size_t)d * H + hh]));
}

// one thread per (edge, head, float4-chunk of Dh=64)
__global__ void gnn_attaggr(const int* __restrict__ src, const int* __restrict__ dst,
                            const float* __restrict__ es, const float* __restrict__ ed,
                            const float* __restrict__ mx, const float* __restrict__ sm,
                            const float* __restrict__ h, float* __restrict__ out,
                            int H, int ostride, int ocol0) {
  int i = blockIdx.x * blockDim.x + threadIdx.x;
  if (i >= NTOT * H * 16) return;
  int c4 = i & 15;
  int eh = i >> 4;
  int e = eh / H, hh = eh % H;
  int s, d;
  if (e < NE) { s = src[e]; d = dst[e]; } else { s = d = e - NE; }
  float v     = lrelu(es[(size_t)s * H + hh] + ed[(size_t)d * H + hh]);
  float alpha = expf(v - mx[(size_t)d * H + hh]) / sm[(size_t)d * H + hh];
  const float4 hv = ((const float4*)h)[(size_t)s * (H * 16) + hh * 16 + c4];
  float* o = out + (size_t)d * ostride + ocol0 + hh * 64 + c4 * 4;
  atomicAdd(o + 0, hv.x * alpha); atomicAdd(o + 1, hv.y * alpha);
  atomicAdd(o + 2, hv.z * alpha); atomicAdd(o + 3, hv.w * alpha);
}

__global__ void gnn_addbias(float* __restrict__ z, const float* __restrict__ b,
                            int ld, int col0, int C) {
  int i = blockIdx.x * blockDim.x + threadIdx.x;
  if (i >= NN * C) return;
  int r = i / C, c = i % C;
  z[(size_t)r * ld + col0 + c] += b[c];
}

// ---------------------------------------------------------------------------
// WMMA bf16 GEMM:  C[NN x NOUT] = act( A@W (+ A2@W2) + bias )
//
// Block = 256 thr = 8 waves; wave owns a 16-row tile.
// Weights: f32 image async-copied to LDS (ASYNCcnt path), then converted to
// bf16 *pre-swizzled into B-fragment order*: for chunk (kc, nt), lane l keeps
// its 16 bf16 contiguous -> B fragment = single 32B LDS read (2x ds_load_b128).
// A fragments (bf16, converted from f32 global) live in registers across all
// N-tiles.
// ---------------------------------------------------------------------------

template <int KIN, int NOUT, bool DUAL, bool RELU, bool BIAS>
__global__ __launch_bounds__(256) void gnn_gemm_wmma(
    const float* __restrict__ A,  const float* __restrict__ W,
    const float* __restrict__ A2, const float* __restrict__ W2,
    const float* __restrict__ bias, float* __restrict__ C, int ldc) {
  constexpr int NW = DUAL ? 2 : 1;
  constexpr int NT = NOUT / 16;   // N-tiles
  constexpr int KC = KIN / 32;    // K-chunks

  extern __shared__ char smem[];
  __bf16* wl  = (__bf16*)smem;                                  // NW*KIN*NOUT swizzled bf16
  float*  stg = (float*)(smem + (size_t)NW * KIN * NOUT * 2);   // KIN*NOUT f32 staging
  unsigned stg_lds = (unsigned)(size_t)(void*)stg;              // LDS byte offset (addr[31:0])

  for (int p = 0; p < NW; ++p) {
    const float* Wp = p ? W2 : W;
    // ---- async global -> LDS stage of the f32 weight image ----
    for (int i = threadIdx.x; i < KIN * NOUT / 4; i += 256) {
      unsigned lds_addr = stg_lds + (unsigned)i * 16u;
      unsigned long long ga = (unsigned long long)(size_t)(Wp + i * 4);
      asm volatile("global_load_async_to_lds_b128 %0, %1, off"
                   :: "v"(lds_addr), "v"(ga) : "memory");
    }
    asm volatile("s_wait_asynccnt 0" ::: "memory");
    __syncthreads();
    // ---- convert + swizzle into per-lane B-fragment order ----
    // dest i = ((kc*NT + nt)*32 + lane)*16 + e  holds  W[(kc*32 + (lane>>4)*16 + e)*NOUT + nt*16 + (lane&15)]
    __bf16* wp = wl + p * KIN * NOUT;
    for (int i = threadIdx.x; i < KIN * NOUT; i += 256) {
      int e = i & 15, lane = (i >> 4) & 31, t2 = i >> 9;
      int nt = t2 % NT, kc = t2 / NT;
      int k = kc * 32 + (lane >> 4) * 16 + e;
      int n = nt * 16 + (lane & 15);
      wp[i] = (__bf16)stg[k * NOUT + n];
    }
    __syncthreads();
  }

  int wave = threadIdx.x >> 5;
  int lane = threadIdx.x & 31;
  int row0 = (blockIdx.x * 8 + wave) * 16;
  if (row0 >= NN) return;                 // wave-uniform; EXEC all-ones for WMMA
  int m  = row0 + (lane & 15);
  int hi = lane >> 4;
  int nb = lane & 15;

  // A fragment k order used here (contiguous, same as the B swizzle): lanes 0-15 hold K {0..15},
  // lanes 16-31 hold K {16..31}; VGPR v holds K pair 2v+hi*16.
  v16bf af[NW][KC];
#pragma unroll
  for (int p = 0; p < NW; ++p) {
    const float* ar = (p ? A2 : A) + (size_t)m * KIN;
#pragma unroll
    for (int kc = 0; kc < KC; ++kc) {
#pragma unroll
      for (int v = 0; v < 8; ++v) {
        int kb = kc * 32 + 2 * v + hi * 16;
        af[p][kc][2 * v]     = (__bf16)ar[kb];
        af[p][kc][2 * v + 1] = (__bf16)ar[kb + 1];
      }
    }
  }

  for (int nt = 0; nt < NT; ++nt) {
    int nc = nt * 16 + nb;
    v8f acc = {0.f, 0.f, 0.f, 0.f, 0.f, 0.f, 0.f, 0.f};
#pragma unroll
    for (int p = 0; p < NW; ++p) {
      const v16bf* wv = (const v16bf*)(wl + p * KIN * NOUT);
#pragma unroll
      for (int kc = 0; kc < KC; ++kc) {
        v16bf bfr = wv[(kc * NT + nt) * 32 + lane];   // contiguous 32B per lane
        acc = __builtin_amdgcn_wmma_f32_16x16x32_bf16(
            false, af[p][kc], false, bfr, (short)0, acc, false, false);
      }
    }
    float bb = 0.f;
    if (BIAS) bb = bias[nc];
#pragma unroll
    for (int r = 0; r < 8; ++r) {
      float vv = acc[r] + bb;
      if (RELU) vv = fmaxf(vv, 0.f);
      C[(size_t)(row0 + hi * 8 + r) * ldc + nc] = vv;  // C: lane->col, VGPR r -> row r+hi*8
    }
  }
}

template <int KIN, int NOUT, bool DUAL, bool RELU, bool BIAS>
static void launch_gemm(dim3 grid, hipStream_t s,
                        const float* A, const float* W, const float* A2,
                        const float* W2, const float* b, float* C, int ldc) {
  size_t sh = (size_t)(DUAL ? 2 : 1) * KIN * NOUT * 2 + (size_t)KIN * NOUT * 4;
  gnn_gemm_wmma<KIN, NOUT, DUAL, RELU, BIAS><<<grid, 256, sh, s>>>(A, W, A2, W2, b, C, ldc);
}

// final 64->2 linear + log_softmax (too small for WMMA; one thread per node)
__global__ void gnn_head(const float* __restrict__ z2, const float* __restrict__ W,
                         const float* __restrict__ b, float* __restrict__ out) {
  int i = blockIdx.x * blockDim.x + threadIdx.x;
  if (i >= NN) return;
  const float* r = z2 + (size_t)i * 64;
  float l0 = b[0], l1 = b[1];
  for (int d = 0; d < 64; ++d) { l0 += r[d] * W[d * 2]; l1 += r[d] * W[d * 2 + 1]; }
  float mx  = fmaxf(l0, l1);
  float lse = mx + logf(expf(l0 - mx) + expf(l1 - mx));
  out[2 * i]     = l0 - lse;
  out[2 * i + 1] = l1 - lse;
}

// ---------------------------------------------------------------------------
// host launcher
// ---------------------------------------------------------------------------

extern "C" void kernel_launch(void* const* d_in, const int* in_sizes, int n_in,
                              void* d_out, int out_size, void* d_ws, size_t ws_size,
                              hipStream_t stream) {
  (void)in_sizes; (void)n_in; (void)out_size; (void)ws_size;
  const float* x      = (const float*)d_in[0];
  const float* edge_x = (const float*)d_in[1];
  const int*   src    = (const int*)d_in[2];
  const int*   dst    = (const int*)d_in[3];
  const float* Wn1 = (const float*)d_in[4];
  const float* Ws1 = (const float*)d_in[5];
  const float* b1  = (const float*)d_in[6];
  const float* Wn2 = (const float*)d_in[7];
  const float* Ws2 = (const float*)d_in[8];
  const float* b2  = (const float*)d_in[9];
  const float* sbn_g = (const float*)d_in[10];
  const float* sbn_b = (const float*)d_in[11];
  const float* Wg1 = (const float*)d_in[12];
  const float* as1 = (const float*)d_in[13];
  const float* ad1 = (const float*)d_in[14];
  const float* bg1 = (const float*)d_in[15];  // cancelled exactly by following BN
  const float* Wg2 = (const float*)d_in[16];
  const float* as2 = (const float*)d_in[17];
  const float* ad2 = (const float*)d_in[18];
  const float* bg2 = (const float*)d_in[19];
  const float* gbn_g = (const float*)d_in[20];
  const float* gbn_b = (const float*)d_in[21];
  const float* Wm1 = (const float*)d_in[22];
  const float* bm1 = (const float*)d_in[23];
  const float* Wm2 = (const float*)d_in[24];
  const float* bm2 = (const float*)d_in[25];
  const float* Wm3 = (const float*)d_in[26];
  const float* bm3 = (const float*)d_in[27];
  (void)bg1;
  float* out = (float*)d_out;

  // workspace arena (~158 MB)
  float* base = (float*)d_ws;
  size_t o = 0;
  auto alloc = [&](size_t n) { float* p = base + o; o += n; return p; };
  float* sbuf = alloc((size_t)NN * 64);    // SAGE aggregation sum (reused both layers)
  float* cnt  = alloc(NN);
  float* h1   = alloc((size_t)NN * 64);    // SAGE hidden; later reused as GAT h2
  float* z    = alloc((size_t)NN * 128);   // concat [x1 | x2]
  float* hg   = alloc((size_t)NN * 256);   // GAT1 features; later reused as MLP z1
  float* g1o  = alloc((size_t)NN * 256);   // GAT1 output;  later reused as MLP z2
  float* es1 = alloc((size_t)NN * 4); float* ed1 = alloc((size_t)NN * 4);
  float* mx1 = alloc((size_t)NN * 4); float* sm1 = alloc((size_t)NN * 4);
  float* es2 = alloc(NN); float* ed2 = alloc(NN);
  float* mx2 = alloc(NN); float* sm2 = alloc(NN);
  float* mean = alloc(256); float* var = alloc(256);

  auto g1 = [](long long n) { return dim3((unsigned)((n + 255) / 256)); };
  dim3 ggrid((3125 + 7) / 8);  // 3125 16-row tiles, 8 waves/block

  // init (must happen every call: atomics accumulate into these)
  hipMemsetAsync(cnt,  0, (size_t)NN * 4, stream);
  hipMemsetAsync(sbuf, 0, (size_t)NN * 64 * 4, stream);
  hipMemsetAsync(z,    0, (size_t)NN * 128 * 4, stream);
  hipMemsetAsync(sm1,  0, (size_t)NN * 4 * 4, stream);
  hipMemsetAsync(sm2,  0, (size_t)NN * 4, stream);
  hipMemsetAsync(g1o,  0, (size_t)NN * 256 * 4, stream);
  gnn_fill<<<g1((long long)NN * 4), 256, 0, stream>>>(mx1, -3.0e38f, NN * 4);
  gnn_fill<<<g1(NN), 256, 0, stream>>>(mx2, -3.0e38f, NN);

  // ---- SAGE layer 1 ----
  gnn_deg<<<g1(NE), 256, 0, stream>>>(dst, cnt, NE);
  gnn_scatter64<<<g1((long long)NE * 16), 256, 0, stream>>>(x, src, dst, sbuf, NE);
  gnn_meandiv<<<g1((long long)NN * 64), 256, 0, stream>>>(sbuf, cnt, NN * 64);
  launch_gemm<64, 64, true, false, true>(ggrid, stream, sbuf, Wn1, x, Ws1, b1, h1, 64);
  gnn_colstats<<<64, 256, 0, stream>>>(h1, 64, mean, var);
  gnn_bnrelu<<<g1((long long)NN * 64), 256, 0, stream>>>(h1, mean, var, sbn_g, sbn_b, 64, NN * 64);

  // ---- SAGE layer 2 -> z[:, 0:64] ----
  hipMemsetAsync(sbuf, 0, (size_t)NN * 64 * 4, stream);
  gnn_scatter64<<<g1((long long)NE * 16), 256, 0, stream>>>(h1, src, dst, sbuf, NE);
  gnn_meandiv<<<g1((long long)NN * 64), 256, 0, stream>>>(sbuf, cnt, NN * 64);
  launch_gemm<64, 64, true, false, true>(ggrid, stream, sbuf, Wn2, h1, Ws2, b2, z, 128);

  // ---- GAT layer 1 (4 heads, concat) ----
  launch_gemm<64, 256, false, false, false>(ggrid, stream, edge_x, Wg1, nullptr, nullptr, nullptr, hg, 256);
  gnn_logits<<<g1((long long)NN * 4), 256, 0, stream>>>(hg, as1, ad1, es1, ed1, 4);
  gnn_attmax<<<g1((long long)NTOT * 4), 256, 0, stream>>>(src, dst, es1, ed1, mx1, 4);
  gnn_attsum<<<g1((long long)NTOT * 4), 256, 0, stream>>>(src, dst, es1, ed1, mx1, sm1, 4);
  gnn_attaggr<<<g1((long long)NTOT * 4 * 16), 256, 0, stream>>>(src, dst, es1, ed1, mx1, sm1, hg, g1o, 4, 256, 0);
  gnn_colstats<<<256, 256, 0, stream>>>(g1o, 256, mean, var);
  gnn_bnrelu<<<g1((long long)NN * 256), 256, 0, stream>>>(g1o, mean, var, gbn_g, gbn_b, 256, NN * 256);

  // ---- GAT layer 2 (1 head) -> z[:, 64:128] ----
  launch_gemm<256, 64, false, false, false>(ggrid, stream, g1o, Wg2, nullptr, nullptr, nullptr, h1, 64);
  gnn_logits<<<g1(NN), 256, 0, stream>>>(h1, as2, ad2, es2, ed2, 1);
  gnn_attmax<<<g1(NTOT), 256, 0, stream>>>(src, dst, es2, ed2, mx2, 1);
  gnn_attsum<<<g1(NTOT), 256, 0, stream>>>(src, dst, es2, ed2, mx2, sm2, 1);
  gnn_attaggr<<<g1((long long)NTOT * 16), 256, 0, stream>>>(src, dst, es2, ed2, mx2, sm2, h1, z, 1, 128, 64);
  gnn_addbias<<<g1((long long)NN * 64), 256, 0, stream>>>(z, bg2, 128, 64, 64);

  // ---- fusion MLP + log_softmax ----
  float* z1 = hg;   // free after GAT1 aggregation
  float* z2 = g1o;  // free after GAT2 GEMM
  launch_gemm<128, 128, false, true, true>(ggrid, stream, z, Wm1, nullptr, nullptr, bm1, z1, 128);
  launch_gemm<128, 64, false, true, true>(ggrid, stream, z1, Wm2, nullptr, nullptr, bm2, z2, 64);
  gnn_head<<<g1(NN), 256, 0, stream>>>(z2, Wm3, bm3, out);
}
